// ActorRNN_72438918414879
// MI455X (gfx1250) — compile-verified, matrix operand order checked by reading, not run on hardware
//
#include <hip/hip_runtime.h>
#include <hip/hip_bf16.h>
#include <stdint.h>

// ---------------------------------------------------------------------------
// ActorRNN forward, fused for gfx1250 (MI455X, wave32, WMMA).
//   T=256, B=512, D=128, H=128, A=8
// Plan:
//   k_stats_partial : deterministic column reduction (sum, sumsq) -> ws partials
//   k_stats_final   : Chan-merge with (mean_obs, welford_S, count) -> M, inv_std
//   k_gru_fused     : 32 blocks x 16 batch rows; all weights f16-transposed in
//                     LDS (~295KB of the 320KB WGP LDS); per timestep:
//                     async global->LDS double-buffered obs staging
//                     (GLOBAL_LOAD_ASYNC_TO_LDS_B128 + s_wait_asynccnt),
//                     hg = h@Wh (WMMA), emb = relu(norm(obs)@Wemb) (WMMA),
//                     xg = emb@Wi (WMMA), GRU gates (VALU exp), act (WMMA),
//                     alpha/beta heads (WMMA, padded to 16 cols), softplus+1.
// ---------------------------------------------------------------------------

#define TT 256
#define BB 512
#define DD 128
#define HH 128
#define AA 8

typedef __attribute__((ext_vector_type(16))) _Float16 v16h;
typedef __attribute__((ext_vector_type(8)))  _Float16 v8h;
typedef __attribute__((ext_vector_type(8)))  float    v8f;

// d_out layout (floats): h_last[512*128] | alpha[256*512*8] | beta[...] | M[128] | S[128]
#define OFF_ALPHA (BB*HH)
#define OFF_BETA  (OFF_ALPHA + TT*BB*AA)
#define OFF_M     (OFF_BETA + TT*BB*AA)
#define OFF_S     (OFF_M + DD)

// ws layout (floats): partials[128 blocks * 256] | M[128] | inv_std[128]
#define WS_M      32768
#define WS_IS     32896

// ---- gfx1250 async global->LDS path (guarded; falls back to sync loads) ---
#if defined(__has_builtin)
#if __has_builtin(__builtin_amdgcn_global_load_async_to_lds_b128)
#define USE_ASYNC_LDS 1
#endif
#endif

#ifdef USE_ASYNC_LDS
// Builtin signature (from clang diagnostic): first param is
//   "__attribute__((__vector_size__(4*sizeof(int)))) int __device__ *"
// i.e. int4* in the global (AS1) address space; LDS side is AS(3).
typedef int v4i_vs __attribute__((vector_size(16)));
typedef __attribute__((address_space(1))) v4i_vs* gasync_p;
typedef __attribute__((address_space(3))) v4i_vs* lasync_p;
// LDS budget: weights+state (286272) + 2 raw obs buffers (16384)
#define SMEM_BYTES 302656
template <int N>
__device__ __forceinline__ void wait_asynccnt() {
#if __has_builtin(__builtin_amdgcn_s_wait_asynccnt)
    __builtin_amdgcn_s_wait_asynccnt(N);
#else
    asm volatile("s_wait_asynccnt %0" ::"n"(N));
#endif
}
#else
#define SMEM_BYTES 286272
#endif

// ---- math helpers ---------------------------------------------------------
__device__ __forceinline__ float fsigmoid(float x) {
    return 1.f / (1.f + __expf(-x));
}
__device__ __forceinline__ float ftanh_fast(float x) {
    x = fminf(fmaxf(x, -15.f), 15.f);
    float e = __expf(-2.f * x);
    return (1.f - e) / (1.f + e);
}
__device__ __forceinline__ float softplus1(float x) {
    return (x > 20.f ? x : __logf(1.f + __expf(x))) + 1.f;
}
__device__ __forceinline__ v8f splat8(float x) {
    v8f v;
#pragma unroll
    for (int i = 0; i < 8; ++i) v[i] = x;
    return v;
}

// Load a 16x32 f16 fragment (A or B operand) from a row-major LDS tile whose
// row pitch is 128 halves.  Lane l holds row (l&15); K sub-chunks per the
// CDNA5 16-bit operand layout.  Two contiguous ds_load_b128 per fragment.
__device__ __forceinline__ v16h ld_frag(const _Float16* __restrict__ base,
                                        int lane, int k0) {
    int row = lane & 15;
    int sel = (lane & 16) ? 8 : 0;
    const _Float16* p = base + row * 128 + k0 + sel;
    v8h lo = *(const v8h*)(p);
    v8h hi = *(const v8h*)(p + 16);
    return __builtin_shufflevector(lo, hi, 0,1,2,3,4,5,6,7,8,9,10,11,12,13,14,15);
}

__device__ __forceinline__ v8f wmma16(v16h a, v16h b, v8f c) {
    // D = A(16x32) x B(32x16) + C, f32 accumulate
    return __builtin_amdgcn_wmma_f32_16x16x32_f16(false, a, false, b,
                                                  (short)0, c, false, false);
}

// ---------------------------------------------------------------------------
// Kernel 1a: per-block partial (sum, sumsq) per feature.  128 blocks x 128 thr.
__global__ void k_stats_partial(const float* __restrict__ obs,
                                float* __restrict__ ws) {
    const int d   = threadIdx.x;
    const int blk = blockIdx.x;
    const int rows = (TT * BB) / 128;     // 1024 rows per block
    size_t base = (size_t)blk * rows * DD + d;
    float s = 0.f, q = 0.f;
    for (int r = 0; r < rows; ++r) {
        float x = obs[base + (size_t)r * DD];
        s += x;
        q += x * x;
    }
    ws[blk * 256 + d]       = s;
    ws[blk * 256 + 128 + d] = q;
}

// Kernel 1b: Chan-merge batch stats with prior running stats.  1 block x 128.
__global__ void k_stats_final(const float* __restrict__ ws_part,
                              const float* __restrict__ mean_obs,
                              const float* __restrict__ welford_S,
                              const int*   __restrict__ run_count,
                              float* __restrict__ ws_M,
                              float* __restrict__ ws_IS,
                              float* __restrict__ out) {
    const int d = threadIdx.x;
    float s = 0.f, q = 0.f;
    for (int b = 0; b < 128; ++b) {
        s += ws_part[b * 256 + d];
        q += ws_part[b * 256 + 128 + d];
    }
    const float n = (float)(TT * BB);
    float mb  = s / n;
    float m2b = q - n * mb * mb;
    if (m2b < 0.f) m2b = 0.f;
    float c0  = (float)run_count[0];
    float M0  = mean_obs[d];
    float S0  = welford_S[d];
    float tot = c0 + n;
    float dl  = mb - M0;
    float M   = M0 + dl * (n / tot);
    float S   = S0 + m2b + dl * dl * (c0 * n / tot);
    float var = S / (tot - 1.f);
    float is  = 1.f / (sqrtf(var) + 1e-8f);
    ws_M[d]  = M;
    ws_IS[d] = is;
    out[OFF_M + d] = M;
    out[OFF_S + d] = S;
}

// ---------------------------------------------------------------------------
// Kernel 2: fused embed + GRU scan + heads.  Grid 32 blocks (16 batch rows
// each), 256 threads (8 waves).  All weights LDS-resident, f16, transposed.
__global__ __launch_bounds__(256)
void k_gru_fused(const float* __restrict__ hidden,
                 const float* __restrict__ obs,
                 const unsigned char* __restrict__ dones,
                 const float* __restrict__ ws_M,
                 const float* __restrict__ ws_IS,
                 const float* __restrict__ W_emb, const float* __restrict__ b_emb,
                 const float* __restrict__ Wi,    const float* __restrict__ Wh,
                 const float* __restrict__ bh,
                 const float* __restrict__ W_act, const float* __restrict__ b_act,
                 const float* __restrict__ W_alpha, const float* __restrict__ b_alpha,
                 const float* __restrict__ W_beta,  const float* __restrict__ b_beta,
                 float* __restrict__ out) {
    extern __shared__ char smem[];
    _Float16* WembT = (_Float16*)smem;          // [128][128]  (n-major, k contiguous)
    _Float16* WiT   = WembT + 128 * 128;        // [384][128]
    _Float16* WhT   = WiT   + 384 * 128;        // [384][128]
    _Float16* WactT = WhT   + 384 * 128;        // [128][128]
    _Float16* WabT  = WactT + 128 * 128;        // [2][16][128]  alpha,beta padded
    _Float16* hsh   = WabT  + 2 * 16 * 128;     // [16][128]  hidden state (f16)
    _Float16* obsh  = hsh   + 16 * 128;         // [16][128]  normalized obs (f16)
    _Float16* embh  = obsh  + 16 * 128;         // [16][128]  emb / act scratch
    float* b_emb_s  = (float*)(embh + 16 * 128);
    float* bh_s     = b_emb_s + 128;            // 384
    float* b_act_s  = bh_s + 384;               // 128
    float* b_ab_s   = b_act_s + 128;            // 16
    float* M_s      = b_ab_s + 16;              // 128
    float* IS_s     = M_s + 128;                // 128
#ifdef USE_ASYNC_LDS
    float* obsraw0  = IS_s + 128;               // [16][128] raw f32, buffer 0
    float* obsraw1  = obsraw0 + 16 * 128;       // buffer 1
#endif

    const int tid  = threadIdx.x;
    const int b0   = blockIdx.x * 16;           // first batch row of this block
    const int lane = tid & 31;
    const int w    = tid >> 5;                  // wave 0..7
    const int cl   = lane & 15;
    const int rbase = (lane >> 4) << 3;         // 0 or 8 (row group within tile)

#ifdef USE_ASYNC_LDS
    // Issue one timestep's raw obs tile (2048 f32 = 512 x b128) as async DMA:
    // 2 GLOBAL_LOAD_ASYNC_TO_LDS_B128 per wave (counts 2 on ASYNCcnt).
    auto issue_async_tile = [&](int t, float* dstraw) {
#pragma unroll
        for (int j = 0; j < 2; ++j) {
            int e = (j * 256 + tid) * 4;        // float index within tile
            int r = e >> 7, c = e & 127;
            const float* g = obs + ((size_t)t * BB + b0 + r) * DD + c;
            float* l = dstraw + e;
            __builtin_amdgcn_global_load_async_to_lds_b128(
                (gasync_p)(uintptr_t)g,
                (lasync_p)(unsigned)(uintptr_t)(void*)l,
                0, 0);
        }
    };
#endif

    // ---- load weights (transposed) + biases + stats into LDS --------------
    for (int idx = tid; idx < 128 * 128; idx += 256) {
        int k = idx >> 7, n = idx & 127;
        WembT[n * 128 + k] = (_Float16)W_emb[idx];
        WactT[n * 128 + k] = (_Float16)W_act[idx];
    }
    for (int idx = tid; idx < 128 * 384; idx += 256) {
        int k = idx / 384, n = idx - k * 384;
        WiT[n * 128 + k] = (_Float16)Wi[idx];
        WhT[n * 128 + k] = (_Float16)Wh[idx];
    }
    for (int idx = tid; idx < 2 * 16 * 128; idx += 256) {
        int head = idx >> 11;
        int rem  = idx & 2047;
        int n = rem >> 7, k = rem & 127;
        float v = 0.f;
        if (n < AA) v = head ? W_beta[k * AA + n] : W_alpha[k * AA + n];
        WabT[idx] = (_Float16)v;
    }
    for (int i = tid; i < 128; i += 256) {
        b_emb_s[i] = b_emb[i];
        b_act_s[i] = b_act[i];
        M_s[i]  = ws_M[i];
        IS_s[i] = ws_IS[i];
    }
    for (int i = tid; i < 384; i += 256) bh_s[i] = bh[i];
    if (tid < 16) b_ab_s[tid] = (tid < 8) ? b_alpha[tid] : b_beta[tid - 8];
    for (int idx = tid; idx < 16 * 128; idx += 256) {
        int r = idx >> 7, c = idx & 127;
        hsh[idx] = (_Float16)hidden[(size_t)(b0 + r) * HH + c];
    }
#ifdef USE_ASYNC_LDS
    issue_async_tile(0, obsraw0);
#endif
    __syncthreads();

    // ---- sequential scan over T ------------------------------------------
    for (int t = 0; t < TT; ++t) {
#ifdef USE_ASYNC_LDS
        // double-buffered async staging: issue t+1, retire t, then normalize
        float* cur = (t & 1) ? obsraw1 : obsraw0;
        float* nxt = (t & 1) ? obsraw0 : obsraw1;
        if (t + 1 < TT) {
            issue_async_tile(t + 1, nxt);
            wait_asynccnt<2>();     // in-order: tile t's 2 ops have retired
        } else {
            wait_asynccnt<0>();
        }
        __syncthreads();            // tile t visible to all waves
        for (int idx = tid; idx < 16 * 128; idx += 256) {
            int r = idx >> 7, c = idx & 127;
            float x = cur[idx];
            obsh[idx] = (_Float16)((x - M_s[c]) * IS_s[c]);
            if (dones[t * BB + b0 + r]) hsh[idx] = (_Float16)0.f;
        }
#else
        for (int idx = tid; idx < 16 * 128; idx += 256) {
            int r = idx >> 7, c = idx & 127;
            int gb = b0 + r;
            float x = obs[((size_t)t * BB + gb) * DD + c];
            obsh[idx] = (_Float16)((x - M_s[c]) * IS_s[c]);
            if (dones[t * BB + gb]) hsh[idx] = (_Float16)0.f;
            if (t + 1 < TT)
                __builtin_prefetch(&obs[((size_t)(t + 1) * BB + gb) * DD + c], 0, 1);
        }
#endif
        __syncthreads();

        // hg = h @ Wh + bh   (3 tiles of [16,16] per wave, K=128)
        v8f hg[3];
#pragma unroll
        for (int g = 0; g < 3; ++g) {
            v8f acc = splat8(bh_s[g * 128 + w * 16 + cl]);
            const _Float16* Bt = WhT + (g * 128 + w * 16) * 128;
#pragma unroll
            for (int k0 = 0; k0 < 128; k0 += 32)
                acc = wmma16(ld_frag(hsh, lane, k0), ld_frag(Bt, lane, k0), acc);
            hg[g] = acc;
        }

        // emb = relu(obs_n @ W_emb + b_emb)
        {
            v8f e = splat8(b_emb_s[w * 16 + cl]);
            const _Float16* Bt = WembT + (w * 16) * 128;
#pragma unroll
            for (int k0 = 0; k0 < 128; k0 += 32)
                e = wmma16(ld_frag(obsh, lane, k0), ld_frag(Bt, lane, k0), e);
#pragma unroll
            for (int i = 0; i < 8; ++i) {
                float v = e[i] > 0.f ? e[i] : 0.f;
                embh[(rbase + i) * 128 + w * 16 + cl] = (_Float16)v;
            }
        }
        __syncthreads();

        // xg = emb @ Wi  (no input bias, flax GRUCell)
        v8f xg[3];
#pragma unroll
        for (int g = 0; g < 3; ++g) {
            v8f acc = splat8(0.f);
            const _Float16* Bt = WiT + (g * 128 + w * 16) * 128;
#pragma unroll
            for (int k0 = 0; k0 < 128; k0 += 32)
                acc = wmma16(ld_frag(embh, lane, k0), ld_frag(Bt, lane, k0), acc);
            xg[g] = acc;
        }

        // GRU gates + state update (each wave owns cols [16w,16w+16))
#pragma unroll
        for (int i = 0; i < 8; ++i) {
            int row = rbase + i;
            int col = w * 16 + cl;
            float hold = (float)hsh[row * 128 + col];
            float r = fsigmoid(xg[0][i] + hg[0][i]);
            float z = fsigmoid(xg[1][i] + hg[1][i]);
            float n = ftanh_fast(xg[2][i] + r * hg[2][i]);
            float hn = (1.f - z) * n + z * hold;
            hsh[row * 128 + col] = (_Float16)hn;
        }
        __syncthreads();

        // act = relu(h_new @ W_act + b_act)  -> reuse embh as act buffer
        {
            v8f a = splat8(b_act_s[w * 16 + cl]);
            const _Float16* Bt = WactT + (w * 16) * 128;
#pragma unroll
            for (int k0 = 0; k0 < 128; k0 += 32)
                a = wmma16(ld_frag(hsh, lane, k0), ld_frag(Bt, lane, k0), a);
#pragma unroll
            for (int i = 0; i < 8; ++i) {
                float v = a[i] > 0.f ? a[i] : 0.f;
                embh[(rbase + i) * 128 + w * 16 + cl] = (_Float16)v;
            }
        }
        __syncthreads();

        // heads: wave 0 -> alpha, wave 1 -> beta (wave-uniform branch)
        if (w < 2) {
            v8f hb = splat8(cl < AA ? b_ab_s[w * AA + cl] : 0.f);
            const _Float16* Bt = WabT + w * 16 * 128;
#pragma unroll
            for (int k0 = 0; k0 < 128; k0 += 32)
                hb = wmma16(ld_frag(embh, lane, k0), ld_frag(Bt, lane, k0), hb);
            if (cl < AA) {
                size_t base = (size_t)OFF_ALPHA + (size_t)w * (TT * BB * AA);
#pragma unroll
                for (int i = 0; i < 8; ++i) {
                    int row = rbase + i;
                    out[base + ((size_t)t * BB + b0 + row) * AA + cl] =
                        softplus1(hb[i]);
                }
            }
        }
        __syncthreads();
    }

    // h_last
    for (int idx = tid; idx < 16 * 128; idx += 256) {
        int r = idx >> 7, c = idx & 127;
        out[(size_t)(b0 + r) * HH + c] = (float)hsh[idx];
    }
}

// ---------------------------------------------------------------------------
extern "C" void kernel_launch(void* const* d_in, const int* in_sizes, int n_in,
                              void* d_out, int out_size, void* d_ws, size_t ws_size,
                              hipStream_t stream) {
    const float* hidden     = (const float*)d_in[0];
    const float* obs        = (const float*)d_in[1];
    const unsigned char* dn = (const unsigned char*)d_in[2];
    const float* mean_obs   = (const float*)d_in[3];
    const float* welford_S  = (const float*)d_in[4];
    const int*   run_count  = (const int*)d_in[5];
    const float* W_emb      = (const float*)d_in[6];
    const float* b_emb      = (const float*)d_in[7];
    const float* Wi         = (const float*)d_in[8];
    const float* Wh         = (const float*)d_in[9];
    const float* bh         = (const float*)d_in[10];
    const float* W_act      = (const float*)d_in[11];
    const float* b_act      = (const float*)d_in[12];
    const float* W_alpha    = (const float*)d_in[13];
    const float* b_alpha    = (const float*)d_in[14];
    const float* W_beta     = (const float*)d_in[15];
    const float* b_beta     = (const float*)d_in[16];
    float* out = (float*)d_out;
    float* ws  = (float*)d_ws;

    k_stats_partial<<<128, 128, 0, stream>>>(obs, ws);
    k_stats_final<<<1, 128, 0, stream>>>(ws, mean_obs, welford_S, run_count,
                                         ws + WS_M, ws + WS_IS, out);
    k_gru_fused<<<BB / 16, 256, SMEM_BYTES, stream>>>(
        hidden, obs, dn, ws + WS_M, ws + WS_IS,
        W_emb, b_emb, Wi, Wh, bh, W_act, b_act,
        W_alpha, b_alpha, W_beta, b_beta, out);
}
